// ConvLayer_15126874816931
// MI455X (gfx1250) — matrix operand order repaired and checked by reading, not compiled, not run on hardware
//
#include <hip/hip_runtime.h>
#include <hip/hip_bf16.h>
#include <math.h>

typedef _Float16 v16h __attribute__((ext_vector_type(16)));
typedef _Float16 v8h  __attribute__((ext_vector_type(8)));
typedef float    v8f  __attribute__((ext_vector_type(8)));

#define NBASIS 20
#define MULIN  16
#define WCOLS  448           // 16*(16+8+4)
#define KPAD   32
#define NW     4             // waves (tiles of 16 edges) per block

__device__ __forceinline__ float soft_unit(float x) {
    return x > 0.f ? __expf(-1.f / x) : 0.f;
}

// ---------------- zero output + counters ----------------
__global__ void init_kernel(float* __restrict__ out, int out_size, float* __restrict__ cnt) {
    int i = blockIdx.x * blockDim.x + threadIdx.x;
    if (i < out_size) out[i] = 0.f;
    if (i < 8) cnt[i] = 0.f;
}

// ---------------- per-graph edge/node counts ----------------
__global__ void count_kernel(const int* __restrict__ batch, const int* __restrict__ esrc,
                             int N, int E, float* __restrict__ cnt) {
    int i = blockIdx.x * blockDim.x + threadIdx.x;
    if (i < E) atomicAdd(&cnt[batch[esrc[i]]], 1.f);       // edges per graph
    if (i < N) atomicAdd(&cnt[4 + batch[i]], 1.f);         // nodes per graph
}

// ---------------- fused edge kernel: geometry + WMMA MLP + scatter ----------------
__global__ __launch_bounds__(32 * NW)
void edge_kernel(const float* __restrict__ pos, const float* __restrict__ f_in,
                 const int* __restrict__ esrc, const int* __restrict__ edst,
                 const float* __restrict__ W1, const float* __restrict__ W2,
                 const float* __restrict__ W3, float* __restrict__ out, int E) {
    // weights transposed to [col][K] f16, K padded 20->32 with zeros
    __shared__ __align__(64) _Float16 sW1t[32 * KPAD];
    __shared__ __align__(64) _Float16 sW2t[32 * KPAD];
    __shared__ __align__(64) _Float16 sW3t[WCOLS * KPAD];
    __shared__ __align__(64) _Float16 sStage[NW][16 * KPAD];   // A-matrix staging (basis / h1 / h2)
    __shared__ float sX[NW][16 * MULIN];                        // gathered f_in[src]
    __shared__ float sSh[NW][16 * 12];                          // sph harmonics (9, padded)
    __shared__ int   sDst[NW][16];

    const int tid = threadIdx.x;

    // ---- cooperative weight preload (f32 -> f16, transpose, pad) ----
    for (int idx = tid; idx < 32 * KPAD; idx += blockDim.x) {
        int c = idx >> 5, k = idx & 31;
        sW1t[idx] = (k < NBASIS && c < 20) ? (_Float16)W1[k * 20 + c] : (_Float16)0.f;
        sW2t[idx] = (k < 20 && c < 20)     ? (_Float16)W2[k * 20 + c] : (_Float16)0.f;
    }
    for (int idx = tid; idx < WCOLS * KPAD; idx += blockDim.x) {
        int c = idx >> 5, k = idx & 31;
        sW3t[idx] = (k < 20) ? (_Float16)W3[k * WCOLS + c] : (_Float16)0.f;
    }
    __syncthreads();

    const int w    = tid >> 5;
    const int lane = tid & 31;
    const int hf   = lane >> 4;   // which half of the wave
    const int N16  = lane & 15;

    const long tbase = ((long)blockIdx.x * NW + w) * 16;
    _Float16* stg = &sStage[w][0];

    // ---- per-edge geometry: one edge per lane (lanes 0..15) ----
    if (lane < 16) {
        long e = tbase + lane;
        bool v = (e < (long)E);
        int s = 0, d = 0;
        float vx = 0.f, vy = 0.f, vz = 0.f;
        if (v) {
            s = esrc[e]; d = edst[e];
            vx = pos[3 * d + 0] - pos[3 * s + 0];
            vy = pos[3 * d + 1] - pos[3 * s + 1];
            vz = pos[3 * d + 2] - pos[3 * s + 2];
        }
        float r  = sqrtf(vx * vx + vy * vy + vz * vz);
        float ir = v ? 1.f / r : 0.f;
        float ux = vx * ir, uy = vy * ir, uz = vz * ir;
        const float s3 = 1.73205081f, s5 = 2.23606798f, s15 = 3.87298335f;
        float* shp = &sSh[w][lane * 12];
        shp[0] = 1.f;
        shp[1] = s3 * ux; shp[2] = s3 * uy; shp[3] = s3 * uz;
        shp[4] = s15 * ux * uz;
        shp[5] = s15 * ux * uy;
        shp[6] = s5 * (uy * uy - 0.5f * (ux * ux + uz * uz));
        shp[7] = s15 * uy * uz;
        shp[8] = 0.5f * s15 * (uz * uz - ux * ux);
        // bump basis, values_i = 8*(i+1)/21, step = 8/21; fold in sqrt(20)
        const float bumpc = 1.14136f * 7.38905610f;   // 1.14136*exp(2)
        const float snb   = 4.47213595f;              // sqrt(20)
        float t = r * (21.f / 8.f);
        for (int i = 0; i < KPAD; ++i) {
            float b = 0.f;
            if (i < NBASIS && v) {
                float diff = t - (float)(i + 1);
                b = bumpc * soft_unit(diff + 1.f) * soft_unit(1.f - diff) * snb;
            }
            stg[lane * KPAD + i] = (_Float16)b;
        }
        float* xp = &sX[w][lane * MULIN];
        for (int u = 0; u < MULIN; ++u) xp[u] = v ? f_in[(long)s * MULIN + u] : 0.f;
        sDst[w][lane] = d;
    }
    __syncthreads();

    // A-fragment loader: row M = N16; elems 0-15 -> K = 16*hf+.
    auto load_afrag = [&]() -> v16h {
        const _Float16* rp = stg + N16 * KPAD;
        v8h lo = *(const v8h*)(rp + hf * 16);
        v8h hi = *(const v8h*)(rp + hf * 16 + 8);
        v16h a;
#pragma unroll
        for (int i = 0; i < 8; ++i) { a[i] = lo[i]; a[i + 8] = hi[i]; }
        return a;
    };

    const float s12 = 0.31622777f;   // sqrt(2)/sqrt(20)

    // ---- stage 1: h1 = relu(basis @ W1) * sqrt(2)/sqrt(20) ----
    {
        v16h a = load_afrag();
#pragma unroll
        for (int c = 0; c < 2; ++c) {
            v16h b = *(const v16h*)(sW1t + (c * 16 + N16) * KPAD + hf * 16);
            v8f acc = {};
            acc = __builtin_amdgcn_wmma_f32_16x16x32_f16(false, a, false, b,
                                                         (short)0, acc, false, false);
#pragma unroll
            for (int rI = 0; rI < 8; ++rI) {
                float h = acc[rI] > 0.f ? acc[rI] * s12 : 0.f;
                stg[(rI + 8 * hf) * KPAD + c * 16 + N16] = (_Float16)h;
            }
        }
    }
    __syncthreads();

    // ---- stage 2: h2 = relu(h1 @ W2) * sqrt(2)/sqrt(20) ----
    {
        v16h a = load_afrag();
#pragma unroll
        for (int c = 0; c < 2; ++c) {
            v16h b = *(const v16h*)(sW2t + (c * 16 + N16) * KPAD + hf * 16);
            v8f acc = {};
            acc = __builtin_amdgcn_wmma_f32_16x16x32_f16(false, a, false, b,
                                                         (short)0, acc, false, false);
#pragma unroll
            for (int rI = 0; rI < 8; ++rI) {
                float h = acc[rI] > 0.f ? acc[rI] * s12 : 0.f;
                stg[(rI + 8 * hf) * KPAD + c * 16 + N16] = (_Float16)h;
            }
        }
    }
    __syncthreads();

    // ---- stage 3: w = h2 @ W3, consumed tile-by-tile into the einsum ----
    float acc0[8], acc1[8], acc2[8];
#pragma unroll
    for (int rI = 0; rI < 8; ++rI) { acc0[rI] = 0.f; acc1[rI] = 0.f; acc2[rI] = 0.f; }

    {
        v16h a = load_afrag();
        const float* xw = &sX[w][0];
#pragma unroll
        for (int t = 0; t < 28; ++t) {
            v16h b = *(const v16h*)(sW3t + (t * 16 + N16) * KPAD + hf * 16);
            v8f d = {};
            d = __builtin_amdgcn_wmma_f32_16x16x32_f16(false, a, false, b,
                                                       (short)0, d, false, false);
            if (t < 16) {                     // w0 region: col = u*16 + v, tile has fixed u = t
                const int u = t;
#pragma unroll
                for (int rI = 0; rI < 8; ++rI)
                    acc0[rI] += xw[(rI + 8 * hf) * MULIN + u] * d[rI];
            } else if (t < 24) {              // w1 region: u = 2*(t-16) + N/8, v = N%8
                const int u = 2 * (t - 16) + (N16 >> 3);
#pragma unroll
                for (int rI = 0; rI < 8; ++rI)
                    acc1[rI] += xw[(rI + 8 * hf) * MULIN + u] * d[rI];
            } else {                          // w2 region: u = 4*(t-24) + N/4, v = N%4
                const int u = 4 * (t - 24) + (N16 >> 2);
#pragma unroll
                for (int rI = 0; rI < 8; ++rI)
                    acc2[rI] += xw[(rI + 8 * hf) * MULIN + u] * d[rI];
            }
        }
    }

    // ---- scatter: msg = [o0*sh0 | o1 (x) sh1..3 | o2 (x) sh4..8] * inv, atomically into f_out[dst]
    const float cs = 0.25f * 0.22360680f;  // 1/sqrt(16) * 1/sqrt(20)
#pragma unroll
    for (int rI = 0; rI < 8; ++rI) {
        const int M = rI + 8 * hf;
        long e = tbase + M;
        if (e < (long)E) {
            const int dn = sDst[w][M];
            const float* shp = &sSh[w][M * 12];
            float* ob = out + (long)dn * 60;
            atomicAdd(ob + N16, acc0[rI] * cs);                 // sh[0] == 1
            float a1 = acc1[rI] * cs; int v1 = N16 & 7;
#pragma unroll
            for (int j = 0; j < 3; ++j)
                atomicAdd(ob + 16 + v1 * 3 + j, a1 * shp[1 + j]);
            float a2 = acc2[rI] * cs; int v2 = N16 & 3;
#pragma unroll
            for (int j = 0; j < 5; ++j)
                atomicAdd(ob + 40 + v2 * 5 + j, a2 * shp[4 + j]);
        }
    }
}

// ---------------- final normalization: f_out / sqrt(n_neigh[batch]) ----------------
__global__ void norm_kernel(float* __restrict__ out, const int* __restrict__ batch,
                            const float* __restrict__ cnt, int N) {
    int i = blockIdx.x * blockDim.x + threadIdx.x;
    if (i >= N * 60) return;
    int n = i / 60;
    int g = batch[n];
    // 1/sqrt(n_neigh) = sqrt(nodes_in_graph / edges_in_graph)
    out[i] *= sqrtf(cnt[4 + g] / cnt[g]);
}

extern "C" void kernel_launch(void* const* d_in, const int* in_sizes, int n_in,
                              void* d_out, int out_size, void* d_ws, size_t ws_size,
                              hipStream_t stream) {
    const float* pos  = (const float*)d_in[0];
    const int*   batch= (const int*)  d_in[1];
    const float* f_in = (const float*)d_in[2];
    const int*   esrc = (const int*)  d_in[3];
    const int*   edst = (const int*)  d_in[4];
    const float* W1   = (const float*)d_in[5];
    const float* W2   = (const float*)d_in[6];
    const float* W3   = (const float*)d_in[7];
    float* out = (float*)d_out;
    float* cnt = (float*)d_ws;     // [0..3]=edges per graph, [4..7]=nodes per graph

    const int N = in_sizes[1];
    const int E = in_sizes[3];

    init_kernel<<<(out_size + 255) / 256, 256, 0, stream>>>(out, out_size, cnt);

    int mx = (N > E) ? N : E;
    count_kernel<<<(mx + 255) / 256, 256, 0, stream>>>(batch, esrc, N, E, cnt);

    int nTiles = (E + 15) / 16;
    int blocks = (nTiles + NW - 1) / NW;
    edge_kernel<<<blocks, 32 * NW, 0, stream>>>(pos, f_in, esrc, edst, W1, W2, W3, out, E);

    norm_kernel<<<(N * 60 + 255) / 256, 256, 0, stream>>>(out, batch, cnt, N);
}
